// StableLSE_22024592294198
// MI455X (gfx1250) — matrix operand order repaired and checked by reading, not compiled, not run on hardware
//
#include <hip/hip_runtime.h>
#include <hip/hip_bf16.h>

#define N_NODES 50000
#define N_EDGES 800000
#define F_IN    1433
#define K1_PAD  1440          // F_IN padded to multiple of 32
#define F_HID   256
#define F_OUT   128
#define K_CLS   7
#define BN_EPS  1e-5f

typedef __attribute__((ext_vector_type(16))) __bf16 v16bf;
typedef __attribute__((ext_vector_type(8)))  __bf16 v8bf;
typedef __attribute__((ext_vector_type(2)))  __bf16 v2bf;
typedef __attribute__((ext_vector_type(8)))  float  v8f;

#if defined(__has_builtin)
#if __has_builtin(__builtin_amdgcn_cvt_pk_bf16_f32)
#define HAVE_PK_BF16 1
#endif
#endif

// pack two f32 -> two bf16 in one u32
__device__ __forceinline__ unsigned pack_bf16x2(float a, float b) {
#ifdef HAVE_PK_BF16
    union { v2bf v; unsigned u; } o;
    o.v = __builtin_amdgcn_cvt_pk_bf16_f32(a, b);
    return o.u;
#else
    union { float f; unsigned u; } x, y; x.f = a; y.f = b;
    unsigned ra = (x.u + 0x7FFFu + ((x.u >> 16) & 1u)) >> 16;   // RNE
    unsigned rb = (y.u + 0x7FFFu + ((y.u >> 16) & 1u)) >> 16;
    return (rb << 16) | (ra & 0xFFFFu);
#endif
}

// ---------------------------------------------------------------- utilities
__global__ void gcn_fzero(float* __restrict__ p, long n) {
    long i = (long)blockIdx.x * blockDim.x + threadIdx.x;
    long stride = (long)gridDim.x * blockDim.x;
    for (; i < n; i += stride) p[i] = 0.f;
}

__global__ void gcn_edge_deg(const int* __restrict__ dst, float* __restrict__ deg, int E) {
    int e = blockIdx.x * blockDim.x + threadIdx.x;
    if (e < E) atomicAdd(&deg[dst[e]], 1.0f);
}

__global__ void gcn_finalize_deg(const float* __restrict__ deg,
                                 float* __restrict__ dinv, float* __restrict__ inv, int n) {
    int i = blockIdx.x * blockDim.x + threadIdx.x;
    if (i < n) {
        float d = deg[i] + 1.0f;          // self-loop
        dinv[i] = rsqrtf(d);
        inv[i]  = 1.0f / d;
    }
}

// W[Kdim x Ncols] f32 row-major  ->  Wt[Ncols x Kpad] bf16 (transposed, zero-padded)
__global__ void gcn_wt_bf16(const float* __restrict__ W, unsigned* __restrict__ Wt,
                            int Kdim, int Ncols, int Kpad) {
    int idx = blockIdx.x * blockDim.x + threadIdx.x;    // over Ncols * Kpad/2
    int kph = Kpad >> 1;
    if (idx >= Ncols * kph) return;
    int n  = idx / kph;
    int kp = idx - n * kph;
    int k0 = 2 * kp;
    float f0 = (k0     < Kdim) ? W[(long)k0 * Ncols + n]       : 0.f;
    float f1 = (k0 + 1 < Kdim) ? W[(long)(k0 + 1) * Ncols + n] : 0.f;
    Wt[idx] = pack_bf16x2(f0, f1);
}

// ---------------------------------------------------------------- WMMA GEMM
// C[M x Ncols] = A[M x K] * Bt^T, Bt = [Ncols x Kpad] bf16 (pre-transposed, padded).
// ABF16=false: A is f32 [M x Kdim], converted to bf16 during staging (branchless).
// ABF16=true : A is packed bf16 [M x Kpad] (u32 pairs), pure copy staging.
// Block tile 128x128, 8 waves, each wave 32x64 = 2x4 grid of 16x16 WMMA tiles.
#define BLK_M 128
#define BLK_N 128
#define BLK_K 32

template <bool ABF16>
__global__ __launch_bounds__(256)
void gcn_gemm_wmma(const void* __restrict__ Aptr, const unsigned* __restrict__ Bt,
                   float* __restrict__ C, int M, int Ncols, int Kdim, int Kpad) {
    // u32 LDS tiles, row stride 20 u32 = 80B (multiple of 16B -> aligned b128)
    __shared__ __align__(16) unsigned AsU[BLK_M][20];   // [m][k/2]
    __shared__ __align__(16) unsigned BsU[BLK_N][20];   // [n][k/2]

    const int tid  = threadIdx.x;
    const int lane = tid & 31;
    const int wave = tid >> 5;
    const int m0 = (wave >> 1) * 32;       // 4 wave-rows  (32 rows each)
    const int n0 = (wave & 1) * 64;        // 2 wave-cols  (64 cols each)
    const int blockM = blockIdx.y * BLK_M;
    const int blockN = blockIdx.x * BLK_N;

    v8f acc[2][4] = {};                    // [m-tile][n-tile]

    for (int k0 = 0; k0 < Kpad; k0 += BLK_K) {
        // ---- stage A tile (128 x 32 bf16)
        if (ABF16) {
            const uint4* Ab = (const uint4*)Aptr;       // row stride Kpad/8 uint4
            #pragma unroll
            for (int i = 0; i < 2; ++i) {
                int li = tid + i * 256;                 // 512 chunks of 16B
                int r = li >> 2, c4 = li & 3;
                long gr = min(blockM + r, M - 1);
                uint4 v = Ab[gr * (long)(Kpad >> 3) + (k0 >> 3) + c4];
                *(uint4*)&AsU[r][c4 * 4] = v;
            }
        } else {
            const float* A = (const float*)Aptr;
            #pragma unroll
            for (int i = 0; i < 8; ++i) {
                int li = tid + i * 256;                 // 2048 u32 slots (128 x 16)
                int r = li >> 4, cp = li & 15;
                long gr = min(blockM + r, M - 1);
                int g0 = k0 + 2 * cp;
                float f0 = A[gr * (long)Kdim + min(g0,     Kdim - 1)];
                float f1 = A[gr * (long)Kdim + min(g0 + 1, Kdim - 1)];
                f0 = (g0     < Kdim) ? f0 : 0.f;
                f1 = (g0 + 1 < Kdim) ? f1 : 0.f;
                AsU[r][cp] = pack_bf16x2(f0, f1);
            }
        }
        // ---- stage B tile (128 x 32 bf16): pure uint4 copy, no guards
        {
            #pragma unroll
            for (int i = 0; i < 2; ++i) {
                int li = tid + i * 256;                 // 512 chunks of 16B
                int r = li >> 2, c4 = li & 3;
                uint4 v = *(const uint4*)&Bt[(long)(blockN + r) * (Kpad >> 1) + (k0 >> 1) + c4 * 4];
                *(uint4*)&BsU[r][c4 * 4] = v;
            }
        }
        __syncthreads();

        // Fragment layouts per CDNA5 ISA (16-bit A 16x32 / B 32x16, wave32)
        const int mr  = lane & 15;
        const int khw = (lane >> 4) << 2;   // A K-offset in u32: 0 / 4  (bf16 0/8)
        const int k2w = (lane >> 4) << 2;   // B K-offset in u32: 0 / 4  (bf16 0/8)
        const int nc  = n0 + (lane & 15);

        v16bf af[2], bfr[4];
        #pragma unroll
        for (int mt = 0; mt < 2; ++mt) {
            v8bf lo = *(const v8bf*)&AsU[m0 + mt * 16 + mr][khw];
            v8bf hi = *(const v8bf*)&AsU[m0 + mt * 16 + mr][khw + 8];
            #pragma unroll
            for (int i = 0; i < 8; ++i) { af[mt][i] = lo[i]; af[mt][8 + i] = hi[i]; }
        }
        #pragma unroll
        for (int nt = 0; nt < 4; ++nt) {
            v8bf lo = *(const v8bf*)&BsU[nc + nt * 16][k2w];
            v8bf hi = *(const v8bf*)&BsU[nc + nt * 16][k2w + 8];
            #pragma unroll
            for (int i = 0; i < 8; ++i) { bfr[nt][i] = lo[i]; bfr[nt][8 + i] = hi[i]; }
        }

        #pragma unroll
        for (int mt = 0; mt < 2; ++mt)
            #pragma unroll
            for (int nt = 0; nt < 4; ++nt)
                acc[mt][nt] = __builtin_amdgcn_wmma_f32_16x16x32_bf16(
                    false, af[mt], false, bfr[nt], (short)0, acc[mt][nt], false, false);

        __syncthreads();
    }

    // C/D layout: VGPR p -> row = p + 8*(lane>=16), col = lane&15
    const int rowoff = m0 + ((lane >> 4) << 3);
    const int cbase  = blockN + n0 + (lane & 15);
    if (blockM + BLK_M <= M) {
        // full tile: unconditional stores, incremental addressing
        #pragma unroll
        for (int mt = 0; mt < 2; ++mt) {
            float* Cp = C + (long)(blockM + rowoff + mt * 16) * Ncols + cbase;
            #pragma unroll
            for (int p = 0; p < 8; ++p) {
                #pragma unroll
                for (int nt = 0; nt < 4; ++nt) Cp[nt * 16] = acc[mt][nt][p];
                Cp += Ncols;
            }
        }
    } else {
        #pragma unroll
        for (int mt = 0; mt < 2; ++mt) {
            int r = blockM + rowoff + mt * 16;
            float* Cp = C + (long)r * Ncols + cbase;
            #pragma unroll
            for (int p = 0; p < 8; ++p) {
                if (r + p < M) {
                    #pragma unroll
                    for (int nt = 0; nt < 4; ++nt) Cp[nt * 16] = acc[mt][nt][p];
                }
                Cp += Ncols;
            }
        }
    }
}

// ------------------------------------------------- edge scatter (atomic f32)
// blockDim = 256; (256 >> logF) edges per block, one lane per feature
__global__ void gcn_scatter(const float* __restrict__ h, float* __restrict__ agg,
                            const int* __restrict__ src, const int* __restrict__ dst,
                            const float* __restrict__ dinv, int logF, int E) {
    int per = 256 >> logF;
    int e = blockIdx.x * per + (threadIdx.x >> logF);
    if (e >= E) return;
    int F = 1 << logF;
    int f = threadIdx.x & (F - 1);
    int s = src[e], d = dst[e];
    float coef = dinv[s] * dinv[d];
    atomicAdd(&agg[(long)d * F + f], h[(long)s * F + f] * coef);
}

// ------ layer-1 epilogue: self-loop + bias + BN + ELU -> packed bf16 (F=256)
__global__ void gcn_post1_bf16(const float* __restrict__ h, const float* __restrict__ agg,
                               unsigned* __restrict__ outbf, const float* __restrict__ inv,
                               const float* __restrict__ bias,
                               const float* __restrict__ bnw, const float* __restrict__ bnb,
                               const float* __restrict__ mean, const float* __restrict__ var,
                               long npairs) {
    long idx2 = (long)blockIdx.x * blockDim.x + threadIdx.x;
    long stride = (long)gridDim.x * blockDim.x;
    for (; idx2 < npairs; idx2 += stride) {
        long i = idx2 >> 7;                 // node (128 pairs per node)
        int  p = (int)(idx2 & 127);
        long base = (i << 8) + 2 * p;
        float iv = inv[i];
        float e01[2];
        #pragma unroll
        for (int j = 0; j < 2; ++j) {
            int f = 2 * p + j;
            float v = agg[base + j] + h[base + j] * iv + bias[f];
            v = (v - mean[f]) * rsqrtf(var[f] + BN_EPS) * bnw[f] + bnb[f];
            e01[j] = (v > 0.f) ? v : expm1f(v);
        }
        outbf[idx2] = pack_bf16x2(e01[0], e01[1]);
    }
}

// ------ layer-2 epilogue: self-loop + bias + BN + ELU -> f32 (F=128) to d_out
__global__ void gcn_post2_f32(const float* __restrict__ h, const float* __restrict__ agg,
                              float* __restrict__ out, const float* __restrict__ inv,
                              const float* __restrict__ bias,
                              const float* __restrict__ bnw, const float* __restrict__ bnb,
                              const float* __restrict__ mean, const float* __restrict__ var,
                              long total) {
    long idx = (long)blockIdx.x * blockDim.x + threadIdx.x;
    long stride = (long)gridDim.x * blockDim.x;
    for (; idx < total; idx += stride) {
        long i = idx >> 7;
        int  f = (int)(idx & 127);
        float v = agg[idx] + h[idx] * inv[i] + bias[f];
        v = (v - mean[f]) * rsqrtf(var[f] + BN_EPS) * bnw[f] + bnb[f];
        out[idx] = (v > 0.f) ? v : expm1f(v);
    }
}

// -------------------------------------------- tiny 128x7 projection + softmax
__global__ void gcn_proj_softmax(const float* __restrict__ h, const float* __restrict__ Wp,
                                 const float* __restrict__ bp, float* __restrict__ z, int n) {
    int i = blockIdx.x * blockDim.x + threadIdx.x;
    if (i >= n) return;
    float acc[K_CLS];
    #pragma unroll
    for (int k = 0; k < K_CLS; ++k) acc[k] = bp[k];
    const float* hr = h + (long)i * F_OUT;
    for (int j = 0; j < F_OUT; ++j) {
        float hv = hr[j];
        #pragma unroll
        for (int k = 0; k < K_CLS; ++k) acc[k] += hv * Wp[j * K_CLS + k];
    }
    float m = acc[0];
    #pragma unroll
    for (int k = 1; k < K_CLS; ++k) m = fmaxf(m, acc[k]);
    float s = 0.f;
    #pragma unroll
    for (int k = 0; k < K_CLS; ++k) { acc[k] = expf(acc[k] - m); s += acc[k]; }
    float rs = 1.f / s;
    #pragma unroll
    for (int k = 0; k < K_CLS; ++k) z[(long)i * K_CLS + k] = acc[k] * rs;
}

// ---------------------------------------------------------------- launcher
extern "C" void kernel_launch(void* const* d_in, const int* in_sizes, int n_in,
                              void* d_out, int out_size, void* d_ws, size_t ws_size,
                              hipStream_t stream) {
    const float* x    = (const float*)d_in[0];
    const int*   ei   = (const int*)  d_in[1];
    const float* W1   = (const float*)d_in[2];
    const float* b1   = (const float*)d_in[3];
    const float* bn1w = (const float*)d_in[4];
    const float* bn1b = (const float*)d_in[5];
    const float* bn1m = (const float*)d_in[6];
    const float* bn1v = (const float*)d_in[7];
    const float* W2   = (const float*)d_in[8];
    const float* b2   = (const float*)d_in[9];
    const float* bn2w = (const float*)d_in[10];
    const float* bn2b = (const float*)d_in[11];
    const float* bn2m = (const float*)d_in[12];
    const float* bn2v = (const float*)d_in[13];
    const float* Wp   = (const float*)d_in[14];
    const float* bp   = (const float*)d_in[15];
    const int* src = ei;
    const int* dst = ei + N_EDGES;

    // ---- workspace layout
    // f32: deg(N) dinv(N) inv(N) h1(N*256) agg1(N*256)
    //      layer-2 reuse of h1 region: h2 = first N*128, agg2 = second N*128
    // u32: a1bf(N*128) W1t(256*720) W2t(128*128)
    float* ws   = (float*)d_ws;
    float* deg  = ws;
    float* dinv = ws + (long)N_NODES;
    float* inv  = ws + 2L * N_NODES;
    float* h1   = ws + 3L * N_NODES;
    float* agg1 = h1 + (long)N_NODES * F_HID;
    float* h2   = h1;
    float* agg2 = h1 + (long)N_NODES * F_OUT;
    unsigned* a1bf = (unsigned*)(agg1 + (long)N_NODES * F_HID);   // N*128 u32
    unsigned* W1t  = a1bf + (long)N_NODES * (F_HID / 2);          // 256*720 u32
    unsigned* W2t  = W1t + (long)F_HID * (K1_PAD / 2);            // 128*128 u32

    float* hout = (float*)d_out;                    // [N,128]
    float* zout = hout + (long)N_NODES * F_OUT;     // [N,7]

    // ---- degree / normalization coefficients
    gcn_fzero<<<512, 256, 0, stream>>>(deg, (long)N_NODES);
    gcn_edge_deg<<<(N_EDGES + 255) / 256, 256, 0, stream>>>(dst, deg, N_EDGES);
    gcn_finalize_deg<<<(N_NODES + 255) / 256, 256, 0, stream>>>(deg, dinv, inv, N_NODES);

    // ---- pre-convert weights: transposed, padded bf16
    gcn_wt_bf16<<<(F_HID * (K1_PAD / 2) + 255) / 256, 256, 0, stream>>>(W1, W1t, F_IN, F_HID, K1_PAD);
    gcn_wt_bf16<<<(F_OUT * (F_HID / 2) + 255) / 256, 256, 0, stream>>>(W2, W2t, F_HID, F_OUT, F_HID);

    // ---- layer 1: h1 = x @ W1   (f32 A, converted in staging)
    dim3 g1(F_HID / BLK_N, (N_NODES + BLK_M - 1) / BLK_M);
    gcn_gemm_wmma<false><<<g1, 256, 0, stream>>>(x, W1t, h1, N_NODES, F_HID, F_IN, K1_PAD);

    gcn_fzero<<<8192, 256, 0, stream>>>(agg1, (long)N_NODES * F_HID);
    gcn_scatter<<<N_EDGES, 256, 0, stream>>>(h1, agg1, src, dst, dinv, 8, N_EDGES);
    gcn_post1_bf16<<<8192, 256, 0, stream>>>(h1, agg1, a1bf, inv, b1,
                                             bn1w, bn1b, bn1m, bn1v,
                                             (long)N_NODES * (F_HID / 2));

    // ---- layer 2: h2 = elu(bn(...)) @ W2   (bf16 A, pure-copy staging)
    dim3 g2(F_OUT / BLK_N, (N_NODES + BLK_M - 1) / BLK_M);
    gcn_gemm_wmma<true><<<g2, 256, 0, stream>>>(a1bf, W2t, h2, N_NODES, F_OUT, F_HID, F_HID);

    gcn_fzero<<<8192, 256, 0, stream>>>(agg2, (long)N_NODES * F_OUT);
    gcn_scatter<<<(N_EDGES + 1) / 2, 256, 0, stream>>>(h2, agg2, src, dst, dinv, 7, N_EDGES);
    gcn_post2_f32<<<8192, 256, 0, stream>>>(h2, agg2, hout, inv, b2,
                                            bn2w, bn2b, bn2m, bn2v,
                                            (long)N_NODES * F_OUT);

    // ---- projection + softmax
    gcn_proj_softmax<<<(N_NODES + 127) / 128, 128, 0, stream>>>(hout, Wp, bp, zout, N_NODES);
}
